// Non_LinearGNN_48464410968235
// MI455X (gfx1250) — compile-verified, matrix operand order checked by reading, not run on hardware
//
#include <hip/hip_runtime.h>
#include <hip/hip_bf16.h>

typedef __attribute__((ext_vector_type(16))) _Float16 v16h;
typedef __attribute__((ext_vector_type(8)))  _Float16 v8h;
typedef __attribute__((ext_vector_type(4)))  _Float16 v4h;
typedef __attribute__((ext_vector_type(8)))  float    v8f;

// ---------------------------------------------------------------------------
// WMMA fragment loader. Row-major source (stride 64 halfs). CDNA5 16-bit A/B
// layout: lane<16: row m=lane, K={kBase+0..7, kBase+16..23};
//         lane>=16: row m=lane-16, K={kBase+8..15, kBase+24..31}.
// ---------------------------------------------------------------------------
__device__ __forceinline__ v16h load_frag(const _Float16* base, int lane,
                                          int kBase) {
  int m  = lane & 15;
  int kb = kBase + ((lane & 16) ? 8 : 0);
  const _Float16* p = base + m * 64 + kb;
  v8h lo = *(const v8h*)(p);
  v8h hi = *(const v8h*)(p + 16);
  v16h r;
#pragma unroll
  for (int i = 0; i < 8; ++i) { r[i] = lo[i]; r[i + 8] = hi[i]; }
  return r;
}

// D(16 x NT*16) = prelu( A(16 x KT*32) @ W^T + bias [+ resid] )
// All shape/flag parameters are compile-time -> branchless epilogue.
template <int NT, int KT, int NBIAS, bool RESID>
__device__ __forceinline__ void wmma_dense(const _Float16* in, _Float16* out,
                                           const _Float16* w, const float* bias,
                                           const _Float16* resid, float alpha,
                                           int lane) {
  v16h a[KT];
#pragma unroll
  for (int kt = 0; kt < KT; ++kt) a[kt] = load_frag(in, lane, kt * 32);
  const int ln = lane & 15;
  const int mBase = (lane & 16) ? 8 : 0;
#pragma unroll
  for (int nt = 0; nt < NT; ++nt) {
    v8f acc = {0.f, 0.f, 0.f, 0.f, 0.f, 0.f, 0.f, 0.f};
#pragma unroll
    for (int kt = 0; kt < KT; ++kt) {
      v16h b = load_frag(w + nt * 16 * 64, lane, kt * 32);
      acc = __builtin_amdgcn_wmma_f32_16x16x32_f16(false, a[kt], false, b,
                                                   (short)0, acc, false, false);
    }
    const int n = nt * 16 + ln;
    float bv = bias[(NBIAS < 64 && n >= NBIAS) ? (NBIAS - 1) : n];
    if (NBIAS < 64) bv = (n < NBIAS) ? bv : 0.0f;   // cndmask, no branch
#pragma unroll
    for (int r = 0; r < 8; ++r) {
      float v = acc[r] + bv;
      if (RESID) v += (float)resid[(mBase + r) * 64 + n];
      v = (v >= 0.0f) ? v : alpha * v;
      out[(mBase + r) * 64 + n] = (_Float16)v;
    }
  }
}

// ---------------------------------------------------------------------------
// Weight conversion: 8 slots of 64x64 f16 (zero padded).
// slot0 xi1w, 1 xi2w, 2 xi3w, 3 xi3aw, 4 r1w(32x32), 5 r2w(8x32),
// slot6 l1w(40x40), 7 l1aw(40x40)
// ---------------------------------------------------------------------------
__global__ void prep_weights_kernel(const float* xi1w, const float* xi2w,
                                    const float* xi3w, const float* xi3aw,
                                    const float* r1w, const float* r2w,
                                    const float* l1w, const float* l1aw,
                                    _Float16* W) {
  for (int idx = blockIdx.x * blockDim.x + threadIdx.x; idx < 8 * 4096;
       idx += gridDim.x * blockDim.x) {
    int slot = idx >> 12, off = idx & 4095;
    int n = off >> 6, k = off & 63;
    float v = 0.0f;
    switch (slot) {
      case 0: v = xi1w[n * 64 + k]; break;
      case 1: v = xi2w[n * 64 + k]; break;
      case 2: v = xi3w[n * 64 + k]; break;
      case 3: v = xi3aw[n * 64 + k]; break;
      case 4: v = (n < 32 && k < 32) ? r1w[n * 32 + k] : 0.0f; break;
      case 5: v = (n < 8  && k < 32) ? r2w[n * 32 + k] : 0.0f; break;
      case 6: v = (n < 40 && k < 40) ? l1w[n * 40 + k] : 0.0f; break;
      case 7: v = (n < 40 && k < 40) ? l1aw[n * 40 + k] : 0.0f; break;
    }
    W[idx] = (_Float16)v;
  }
}

// ---------------------------------------------------------------------------
// Edge kernel: one wave = 16 edges. Gathers X = [featT[node], featT[neis]],
// runs the xi MLP (A matrices) and the r MLP (b vectors), via WMMA except the final 8x8 r layers (VALU).
// ---------------------------------------------------------------------------
#define WAVES_EDGE 4
__global__ __launch_bounds__(128) void edge_kernel(
    const int* __restrict__ Xn, const int* __restrict__ Xs,
    const float* __restrict__ fm, const _Float16* __restrict__ W,
    const float* xi1b, const float* xi2b, const float* xi3b,
    const float* xi3ab, const float* xia, const float* r1b, const float* r2b,
    const float* r3aw, const float* r3ab, const float* ra,
    _Float16* __restrict__ Ae, float* __restrict__ be, int V, int E) {
  __shared__ _Float16 lds[WAVES_EDGE][4][16 * 64];
  int wave = threadIdx.x >> 5, lane = threadIdx.x & 31;
  int tile = blockIdx.x * WAVES_EDGE + wave;
  int e0 = tile * 16;
  if (e0 >= E) return;  // wave-uniform

  _Float16* B0 = lds[wave][0];
  _Float16* B1 = lds[wave][1];
  _Float16* B2 = lds[wave][2];
  _Float16* B3 = lds[wave][3];
  const float aXi = xia[0], aR = ra[0];

  // Prefetch the (L2-hot) weight block: 64KB, one probe per 128B line.
  for (int p = threadIdx.x; p < 512; p += 128)
    __builtin_prefetch(W + p * 64, 0, 3);

  // Gather: lanes 0-15 -> node features (cols 0..31) of edge `lane`,
  //         lanes 16-31 -> neighbor features (cols 32..63) of edge `lane-16`.
  {
    int el = e0 + (lane & 15);
    if (el >= E) el = E - 1;
    const int vIdx = (lane < 16) ? Xn[el] : Xs[el];
    const int cOff = (lane < 16) ? 0 : 32;
    const int mRow = lane & 15;
#pragma unroll
    for (int cc = 0; cc < 32; cc += 8) {
      v8h pack;
#pragma unroll
      for (int i = 0; i < 8; ++i)
        pack[i] = (_Float16)fm[(size_t)(cc + i) * V + vIdx];
      *(v8h*)(B0 + mRow * 64 + cOff + cc) = pack;   // ds_store_b128
    }
  }

  // --- xi branch: h = X; 5 residual layers ---
  wmma_dense<4, 2, 64, true >(B0, B1, W + 0,     xi1b,  B0, aXi, lane);
  wmma_dense<4, 2, 64, true >(B1, B2, W + 0,     xi1b,  B0, aXi, lane);
  wmma_dense<4, 2, 64, true >(B2, B1, W + 0,     xi1b,  B0, aXi, lane);
  wmma_dense<4, 2, 64, true >(B1, B2, W + 0,     xi1b,  B0, aXi, lane);
  wmma_dense<4, 2, 64, true >(B2, B1, W + 0,     xi1b,  B0, aXi, lane);
  // f2 = prelu(lin(h, xi2)) -> B2
  wmma_dense<4, 2, 64, false>(B1, B2, W + 4096,  xi2b,  B0, aXi, lane);
  // 5 x { h = prelu(lin(h,xi3)); h = prelu(lin(h,xi3a)+f2) }, f2 fixed in B2
  wmma_dense<4, 2, 64, false>(B2, B1, W + 8192,  xi3b,  B0, aXi, lane);
  wmma_dense<4, 2, 64, true >(B1, B3, W + 12288, xi3ab, B2, aXi, lane);
#pragma unroll 1
  for (int it = 0; it < 4; ++it) {
    wmma_dense<4, 2, 64, false>(B3, B1, W + 8192,  xi3b,  B0, aXi, lane);
    wmma_dense<4, 2, 64, true >(B1, B3, W + 12288, xi3ab, B2, aXi, lane);
  }
  // A = (MU/S/DEG) * h, stored f16 edge-major [e][i*8+j], vectorized b128.
  {
    const float sc = 0.8f / 8.0f / 8.0f;
#pragma unroll
    for (int chunk = lane; chunk < 128; chunk += 32) {
      int m = chunk >> 3, j0 = (chunk & 7) * 8;
      v8h d = *(const v8h*)(B3 + m * 64 + j0);
      v8h o;
#pragma unroll
      for (int i = 0; i < 8; ++i) o[i] = (_Float16)(sc * (float)d[i]);
      if (e0 + m < E) *(v8h*)(Ae + (size_t)(e0 + m) * 64 + j0) = o;
    }
  }

  // --- r branch: xh = X[:, :32] in B1; zero-pad high halves of B1/B2/B3 ---
  {
    v8h z = {};
#pragma unroll
    for (int chunk = lane; chunk < 128; chunk += 32) {
      int m = chunk >> 3, c0 = (chunk & 7) * 8;
      if (c0 < 32) {
        *(v8h*)(B1 + m * 64 + c0) = *(const v8h*)(B0 + m * 64 + c0);
      } else {
        *(v8h*)(B1 + m * 64 + c0) = z;
        *(v8h*)(B2 + m * 64 + c0) = z;
        *(v8h*)(B3 + m * 64 + c0) = z;
      }
    }
  }
  wmma_dense<2, 1, 32, true >(B1, B2, W + 16384, r1b, B1, aR, lane);
  wmma_dense<2, 1, 32, true >(B2, B3, W + 16384, r1b, B1, aR, lane);
  wmma_dense<2, 1, 32, true >(B3, B2, W + 16384, r1b, B1, aR, lane);
  wmma_dense<2, 1, 32, true >(B2, B3, W + 16384, r1b, B1, aR, lane);
  wmma_dense<2, 1, 32, true >(B3, B2, W + 16384, r1b, B1, aR, lane);
  // f2 = prelu(lin(h, r2)) : 32->8, cols 0..7 of B3
  wmma_dense<1, 1, 8, false>(B2, B3, W + 20480, r2b, B1, aR, lane);

  // 5 x 8x8 residual layers in VALU (lanes 0..15, one edge each)
  if (lane < 16 && e0 + lane < E) {
    float f2v[8], h[8], t[8];
#pragma unroll
    for (int i = 0; i < 8; ++i) { f2v[i] = (float)B3[lane * 64 + i]; h[i] = f2v[i]; }
#pragma unroll 1
    for (int it = 0; it < 5; ++it) {
#pragma unroll
      for (int i = 0; i < 8; ++i) {
        float s = r3ab[i] + f2v[i];
#pragma unroll
        for (int j = 0; j < 8; ++j) s += r3aw[i * 8 + j] * h[j];
        t[i] = (s >= 0.0f) ? s : aR * s;
      }
#pragma unroll
      for (int i = 0; i < 8; ++i) h[i] = t[i];
    }
#pragma unroll
    for (int i = 0; i < 8; ++i) be[(size_t)(e0 + lane) * 8 + i] = h[i];
  }
}

// ---------------------------------------------------------------------------
// Message passing: msg = A[e] @ H[neis[e]] + b[e]; segment-sum by node.
// ---------------------------------------------------------------------------
__global__ void msg_kernel(const _Float16* __restrict__ Ae,
                           const float* __restrict__ be,
                           const float* __restrict__ Hin,
                           const int* __restrict__ Xn,
                           const int* __restrict__ Xs,
                           float* __restrict__ Hout, int E) {
  int e = blockIdx.x * blockDim.x + threadIdx.x;
  if (e >= E) return;
  int nb = Xs[e], nd = Xn[e];
  float h[8];
#pragma unroll
  for (int j = 0; j < 8; ++j) h[j] = Hin[(size_t)nb * 8 + j];
#pragma unroll
  for (int i = 0; i < 8; ++i) {
    v8h arow = *(const v8h*)(Ae + (size_t)e * 64 + i * 8);
    float s = be[(size_t)e * 8 + i];
#pragma unroll
    for (int j = 0; j < 8; ++j) s += (float)arow[j] * h[j];
    atomicAdd(&Hout[(size_t)nd * 8 + i], s);
  }
}

// out = concat(featT, H) -> (V,64) zero padded f32; also seed o buffer.
__global__ void build_out_kernel(const float* __restrict__ fm,
                                 const float* __restrict__ H,
                                 float* __restrict__ out64,
                                 float* __restrict__ oBuf, int V) {
  int v = blockIdx.x * blockDim.x + threadIdx.x;
  if (v >= V) return;
  for (int c = 0; c < 32; ++c) {
    float x = fm[(size_t)c * V + v];
    out64[(size_t)v * 64 + c] = x; oBuf[(size_t)v * 64 + c] = x;
  }
#pragma unroll
  for (int i = 0; i < 8; ++i) {
    float x = H[(size_t)v * 8 + i];
    out64[(size_t)v * 64 + 32 + i] = x; oBuf[(size_t)v * 64 + 32 + i] = x;
  }
#pragma unroll
  for (int c = 40; c < 64; c += 4) {
    float4 z = {0.f, 0.f, 0.f, 0.f};
    *(float4*)(out64 + (size_t)v * 64 + c) = z;
    *(float4*)(oBuf + (size_t)v * 64 + c) = z;
  }
}

// ---------------------------------------------------------------------------
// BN-block GEMM: o' = prelu(lin1a(prelu(lin1(o))) + out); accumulate BN stats.
// ---------------------------------------------------------------------------
#define WAVES_BN 4
__global__ __launch_bounds__(128) void bn_gemm_kernel(
    const float* __restrict__ oIn, const float* __restrict__ out64,
    const _Float16* __restrict__ W, const float* l1b, const float* l1ab,
    const float* ga, float* __restrict__ oOut, float* __restrict__ stats,
    int V) {
  __shared__ _Float16 lds[WAVES_BN][2][16 * 64];
  int wave = threadIdx.x >> 5, lane = threadIdx.x & 31;
  int tile = blockIdx.x * WAVES_BN + wave;
  int r0 = tile * 16;
  if (r0 >= V) return;  // wave-uniform
  _Float16* T0 = lds[wave][0];
  _Float16* T1 = lds[wave][1];
  const float alpha = ga[0];

  // Load 16x64 f32 tile -> f16 LDS (b128 global loads, b64 LDS stores).
#pragma unroll
  for (int chunk = lane; chunk < 256; chunk += 32) {
    int m = chunk >> 4, c0 = (chunk & 15) * 4;
    int row = r0 + m; if (row >= V) row = V - 1;
    float4 x = *(const float4*)(oIn + (size_t)row * 64 + c0);
    v4h h = {(_Float16)x.x, (_Float16)x.y, (_Float16)x.z, (_Float16)x.w};
    *(v4h*)(T0 + m * 64 + c0) = h;
  }
  // layer 1: prelu(o @ l1w^T + l1b)
  wmma_dense<4, 2, 40, false>(T0, T1, W + 24576, l1b, T0, alpha, lane);
  // layer 2 + residual(global f32) + prelu + store + stats
  {
    v16h a[2];
#pragma unroll
    for (int kt = 0; kt < 2; ++kt) a[kt] = load_frag(T1, lane, kt * 32);
    const int ln = lane & 15;
    const int mBase = (lane & 16) ? 8 : 0;
#pragma unroll
    for (int nt = 0; nt < 4; ++nt) {
      v8f acc = {0.f, 0.f, 0.f, 0.f, 0.f, 0.f, 0.f, 0.f};
#pragma unroll
      for (int kt = 0; kt < 2; ++kt) {
        v16h b = load_frag(W + 28672 + nt * 16 * 64, lane, kt * 32);
        acc = __builtin_amdgcn_wmma_f32_16x16x32_f16(false, a[kt], false, b,
                                                     (short)0, acc, false, false);
      }
      const int n = nt * 16 + ln;
      float bv = l1ab[(n >= 40) ? 39 : n];
      bv = (n < 40) ? bv : 0.0f;
      float s = 0.0f, sq = 0.0f;
#pragma unroll
      for (int r = 0; r < 8; ++r) {
        int row = r0 + mBase + r;
        int rowc = (row >= V) ? (V - 1) : row;
        float v = acc[r] + bv + out64[(size_t)rowc * 64 + n];
        v = (v >= 0.0f) ? v : alpha * v;
        if (row < V) {
          oOut[(size_t)row * 64 + n] = v;
          s += v; sq += v * v;
        }
      }
      if (n < 40) {
        atomicAdd(&stats[n], s);
        atomicAdd(&stats[64 + n], sq);
      }
    }
  }
}

__global__ void bn_norm_kernel(const float* __restrict__ oPre,
                               float* __restrict__ oNext,
                               const float* __restrict__ stats,
                               const float* __restrict__ bn_g,
                               const float* __restrict__ bn_b, int V) {
  int v = blockIdx.x * blockDim.x + threadIdx.x;
  if (v >= V) return;
  const float invV = 1.0f / (float)V;
  for (int c0 = 0; c0 < 40; c0 += 4) {
    float4 x = *(const float4*)(oPre + (size_t)v * 64 + c0);
    float o[4] = {x.x, x.y, x.z, x.w};
    float4 r;
    float* rp = &r.x;
#pragma unroll
    for (int i = 0; i < 4; ++i) {
      int c = c0 + i;
      float mean = stats[c] * invV;
      float var  = stats[64 + c] * invV - mean * mean;
      rp[i] = (o[i] - mean) * rsqrtf(var + 1e-5f) * bn_g[c] + bn_b[c];
    }
    *(float4*)(oNext + (size_t)v * 64 + c0) = r;
  }
#pragma unroll
  for (int c = 40; c < 64; c += 4) {
    float4 z = {0.f, 0.f, 0.f, 0.f};
    *(float4*)(oNext + (size_t)v * 64 + c) = z;
  }
}

__global__ void final_kernel(const float* __restrict__ oA, const float* l2w,
                             const float* l2b, const float* l3w,
                             const float* l3b, const float* l3aw,
                             const float* l3ab, const float* ga,
                             float* __restrict__ outp, int V) {
  int v = blockIdx.x * blockDim.x + threadIdx.x;
  if (v >= V) return;
  float a = ga[0];
  float o0 = l2b[0], o1 = l2b[1];
  for (int c = 0; c < 40; ++c) {
    float x = oA[(size_t)v * 64 + c];
    o0 += l2w[c] * x;
    o1 += l2w[40 + c] * x;
  }
  float p0 = (o0 >= 0.0f) ? o0 : a * o0;
  float p1 = (o1 >= 0.0f) ? o1 : a * o1;
  float q0 = p0, q1 = p1;
#pragma unroll 1
  for (int it = 0; it < 10; ++it) {
    float t0 = l3b[0] + l3w[0] * q0 + l3w[1] * q1;
    float t1 = l3b[1] + l3w[2] * q0 + l3w[3] * q1;
    t0 = (t0 >= 0.0f) ? t0 : a * t0;
    t1 = (t1 >= 0.0f) ? t1 : a * t1;
    float u0 = l3ab[0] + l3aw[0] * t0 + l3aw[1] * t1 + p0;
    float u1 = l3ab[1] + l3aw[2] * t0 + l3aw[3] * t1 + p1;
    q0 = (u0 >= 0.0f) ? u0 : a * u0;
    q1 = (u1 >= 0.0f) ? u1 : a * u1;
  }
  outp[v] = q0;
  outp[(size_t)V + v] = q1;
}

// ---------------------------------------------------------------------------
extern "C" void kernel_launch(void* const* d_in, const int* in_sizes, int n_in,
                              void* d_out, int out_size, void* d_ws,
                              size_t ws_size, hipStream_t stream) {
  const int*   Xn    = (const int*)d_in[0];
  const int*   Xs    = (const int*)d_in[1];
  const float* fm    = (const float*)d_in[2];
  const float* ns    = (const float*)d_in[3];
  const float* xi1w  = (const float*)d_in[4];
  const float* xi1b  = (const float*)d_in[5];
  const float* xi2w  = (const float*)d_in[6];
  const float* xi2b  = (const float*)d_in[7];
  const float* xi3w  = (const float*)d_in[8];
  const float* xi3b  = (const float*)d_in[9];
  const float* xi3aw = (const float*)d_in[10];
  const float* xi3ab = (const float*)d_in[11];
  const float* xia   = (const float*)d_in[12];
  const float* r1w   = (const float*)d_in[13];
  const float* r1b   = (const float*)d_in[14];
  const float* r2w   = (const float*)d_in[15];
  const float* r2b   = (const float*)d_in[16];
  const float* r3aw  = (const float*)d_in[17];
  const float* r3ab  = (const float*)d_in[18];
  const float* ra    = (const float*)d_in[19];
  const float* l1w   = (const float*)d_in[20];
  const float* l1b   = (const float*)d_in[21];
  const float* l1aw  = (const float*)d_in[22];
  const float* l1ab  = (const float*)d_in[23];
  const float* l2w   = (const float*)d_in[24];
  const float* l2b   = (const float*)d_in[25];
  const float* l3w   = (const float*)d_in[26];
  const float* l3b   = (const float*)d_in[27];
  const float* l3aw  = (const float*)d_in[28];
  const float* l3ab  = (const float*)d_in[29];
  const float* ga    = (const float*)d_in[30];
  const float* bn_g  = (const float*)d_in[31];
  const float* bn_b  = (const float*)d_in[32];
  float* outp = (float*)d_out;

  const int E = in_sizes[0];
  const int V = in_sizes[3] / 8;

  char* ws = (char*)d_ws;
  size_t off = 0;
  auto alloc = [&](size_t bytes) -> void* {
    void* p = ws + off;
    off = (off + bytes + 255) & ~(size_t)255;
    return p;
  };
  _Float16* W     = (_Float16*)alloc((size_t)8 * 4096 * 2);
  _Float16* Ae    = (_Float16*)alloc((size_t)E * 64 * 2);
  float*    be    = (float*)alloc((size_t)E * 8 * 4);
  float*    H0    = (float*)alloc((size_t)V * 8 * 4);
  float*    H1    = (float*)alloc((size_t)V * 8 * 4);
  float*    out64 = (float*)alloc((size_t)V * 64 * 4);
  float*    oA    = (float*)alloc((size_t)V * 64 * 4);
  float*    oB    = (float*)alloc((size_t)V * 64 * 4);
  float*    stats = (float*)alloc((size_t)10 * 128 * 4);

  hipMemsetAsync(stats, 0, (size_t)10 * 128 * 4, stream);
  hipMemcpyAsync(H0, ns, (size_t)V * 8 * 4, hipMemcpyDeviceToDevice, stream);

  prep_weights_kernel<<<32, 256, 0, stream>>>(xi1w, xi2w, xi3w, xi3aw, r1w,
                                              r2w, l1w, l1aw, W);

  int eTiles = (E + 15) / 16;
  edge_kernel<<<(eTiles + WAVES_EDGE - 1) / WAVES_EDGE, 128, 0, stream>>>(
      Xn, Xs, fm, W, xi1b, xi2b, xi3b, xi3ab, xia, r1b, r2b, r3aw, r3ab, ra,
      Ae, be, V, E);

  hipMemsetAsync(H1, 0, (size_t)V * 8 * 4, stream);
  msg_kernel<<<(E + 255) / 256, 256, 0, stream>>>(Ae, be, H0, Xn, Xs, H1, E);
  hipMemsetAsync(H0, 0, (size_t)V * 8 * 4, stream);
  msg_kernel<<<(E + 255) / 256, 256, 0, stream>>>(Ae, be, H1, Xn, Xs, H0, E);

  build_out_kernel<<<(V + 255) / 256, 256, 0, stream>>>(fm, H0, out64, oA, V);

  int vTiles = (V + 15) / 16;
  for (int it = 0; it < 10; ++it) {
    bn_gemm_kernel<<<(vTiles + WAVES_BN - 1) / WAVES_BN, 128, 0, stream>>>(
        oA, out64, W, l1b, l1ab, ga, oB, stats + it * 128, V);
    bn_norm_kernel<<<(V + 255) / 256, 256, 0, stream>>>(oB, oA, stats + it * 128,
                                                        bn_g, bn_b, V);
  }

  final_kernel<<<(V + 255) / 256, 256, 0, stream>>>(oA, l2w, l2b, l3w, l3b,
                                                    l3aw, l3ab, ga, outp, V);
}
